// RNNLM_57329223467355
// MI455X (gfx1250) — matrix operand order repaired and checked by reading, not compiled, not run on hardware
//
#include <hip/hip_runtime.h>
#include <math.h>

// ---------- WMMA types ----------
typedef __attribute__((ext_vector_type(16))) __bf16    v16bf;
typedef __attribute__((ext_vector_type(8)))  float     v8f;
typedef __attribute__((ext_vector_type(4)))  unsigned  uv4;
typedef int vsi4 __attribute__((vector_size(4 * sizeof(int))));

union AF  { unsigned u[8]; v16bf v; };
union AFQ { uv4 q[2]; unsigned u[8]; v16bf v; };

#if defined(__has_builtin)
#if __has_builtin(__builtin_amdgcn_global_load_async_to_lds_b128)
#define HAVE_ASYNC_LDS 1
#endif
#endif
#ifndef HAVE_ASYNC_LDS
#define HAVE_ASYNC_LDS 0
#endif

// native bf16 convert (RNE) -> v_cvt_pk_bf16_f32 on gfx1250
static __device__ __forceinline__ unsigned short f2bf(float x) {
    __bf16 b = (__bf16)x;
    return __builtin_bit_cast(unsigned short, b);
}
static __device__ __forceinline__ unsigned pack2(float lo, float hi) {
    return (unsigned)f2bf(lo) | ((unsigned)f2bf(hi) << 16);
}

static __device__ __forceinline__ v8f wmma_bf16(v16bf a, v16bf b, v8f c) {
    return __builtin_amdgcn_wmma_f32_16x16x32_bf16(
        false, a, false, b, (short)0, c, false, false);
}

#if HAVE_ASYNC_LDS
static __device__ __forceinline__ void async_cp16(const unsigned* g, unsigned* l) {
    __builtin_amdgcn_global_load_async_to_lds_b128(
        (__attribute__((address_space(1))) vsi4*)g,
        (__attribute__((address_space(3))) vsi4*)l, 0, 0);
}
#endif
static __device__ __forceinline__ void wait_async0() {
#if defined(__has_builtin)
#if __has_builtin(__builtin_amdgcn_s_wait_asynccnt)
    __builtin_amdgcn_s_wait_asynccnt(0);
#else
    asm volatile("s_wait_asynccnt 0" ::: "memory");
#endif
#else
    asm volatile("s_wait_asynccnt 0" ::: "memory");
#endif
}

// A fragment (16x32 bf16) from a row-major bf16 matrix. lda in elements (even).
// ISA layout: lane(r,half); VGPR p holds K = (p<4 ? 2p : 2p+8) + 8*half, K+1.
static __device__ __forceinline__ v16bf load_a_rowmajor(
        const unsigned short* __restrict__ A, int lda, int k0) {
    int lane = threadIdx.x & 31, lh = lane >> 4, r = lane & 15;
    const unsigned* row = reinterpret_cast<const unsigned*>(A + (size_t)r * lda);
    AF f;
#pragma unroll
    for (int p = 0; p < 8; ++p) {
        int k = ((p < 4) ? (2 * p) : (2 * p + 8)) + 8 * lh + k0;
        f.u[p] = row[k >> 1];
    }
    return f.v;
}

// B fragment (32x16 bf16) from pair-packed u32 array: word[kpair*N + n] = {K=2kp, K=2kp+1}.
static __device__ __forceinline__ v16bf load_b_packed(
        const unsigned* __restrict__ Bpk, int N, int kpair0, int ncol0) {
    int lane = threadIdx.x & 31, lh = lane >> 4, n = lane & 15;
    AF f;
#pragma unroll
    for (int j = 0; j < 8; ++j) {
        int kp = kpair0 + ((j < 4) ? j : (j + 4)) + 4 * lh;
        f.u[j] = Bpk[(size_t)kp * N + ncol0 + n];
    }
    return f.v;
}

// ---------- constants ----------
#define SEQ   128
#define BATCH 32
#define HID   1024
#define EMB   1024
#define VOCAB 32000
#define MTOT  (SEQ * BATCH)  // 4096
#define LApad 20             // lA row stride in u32: 16B-aligned, bank-conflict-free

// ---------- kernel: pack fp32 KxN weight into pair-interleaved bf16 ----------
__global__ void rnnlm_pack_w(const float* __restrict__ W, unsigned* __restrict__ out,
                             int K, int N) {
    int total = (K >> 1) * N;
    for (int i = blockIdx.x * blockDim.x + threadIdx.x; i < total;
         i += gridDim.x * blockDim.x) {
        int kp = i / N, n = i - kp * N;
        float lo = W[(size_t)(2 * kp) * N + n];
        float hi = W[(size_t)(2 * kp + 1) * N + n];
        out[i] = pack2(lo, hi);
    }
}

// ---------- kernel: fp32 -> bf16 elementwise ----------
__global__ void rnnlm_cvt_bf16(const float* __restrict__ in,
                               unsigned short* __restrict__ out, int ntot) {
    for (int i = blockIdx.x * blockDim.x + threadIdx.x; i < ntot;
         i += gridDim.x * blockDim.x)
        out[i] = f2bf(in[i]);
}

// ---------- kernel: ih1_all = gather(emb, x) @ W_ih1   (fp32 out) ----------
__global__ __launch_bounds__(256) void rnnlm_gemm_ih1(
        const int* __restrict__ x, const float* __restrict__ emb,
        const unsigned* __restrict__ Wpk, float* __restrict__ out) {
    int id = blockIdx.x * 8 + (threadIdx.x >> 5);     // 0..16383
    int mt = id >> 6, nt = id & 63;
    int lane = threadIdx.x & 31, lh = lane >> 4, r = lane & 15;
    int token = x[mt * 16 + r];
    const float* arow = emb + (size_t)token * EMB;
    v8f c = {};
    for (int k = 0; k < EMB; k += 32) {
        AF fa;
#pragma unroll
        for (int p = 0; p < 8; ++p) {
            int kk = ((p < 4) ? (2 * p) : (2 * p + 8)) + 8 * lh + k;
            fa.u[p] = pack2(arow[kk], arow[kk + 1]);
        }
        v16bf b = load_b_packed(Wpk, HID, k >> 1, nt * 16);
        c = wmma_bf16(fa.v, b, c);
    }
#pragma unroll
    for (int vi = 0; vi < 8; ++vi) {
        int row = mt * 16 + vi + 8 * lh;
        out[(size_t)row * HID + nt * 16 + r] = c[vi];
    }
}

// ---------- kernel: one recurrence step ----------
//   kind 0: h2_t = tanh(h1_t @ W_ih2 + h2_{t-1} @ W_hh2 + b2)
//   kind 1: h1_{t+1} = tanh(ih1_slice + h1_t @ W_hh1 + b1)
__global__ __launch_bounds__(256) void rnnlm_step(
        int do_h2, int do_h1,
        const unsigned short* __restrict__ h1cur,
        const unsigned short* __restrict__ h2prev,
        const unsigned* __restrict__ Wpk_hh1,
        const unsigned* __restrict__ Wpk_ih2,
        const unsigned* __restrict__ Wpk_hh2,
        const float* __restrict__ ih1_slice,
        const float* __restrict__ b1, const float* __restrict__ b2,
        unsigned short* __restrict__ h1next_bf, float* __restrict__ h1last,
        unsigned short* __restrict__ h2out_bf, float* __restrict__ h2last) {
    int id = blockIdx.x * 8 + (threadIdx.x >> 5);   // 0..255
    int kind = id >> 7;                             // 0: h2, 1: h1_{t+1}
    int tix = id & 127;
    int mt = tix >> 6, nt = tix & 63;               // mt 0..1, nt 0..63
    int lane = threadIdx.x & 31, lh = lane >> 4, r = lane & 15;
    int ncol = nt * 16 + r;

    if (kind == 0) {
        if (!do_h2) return;
        const unsigned short* A1 = h1cur + (size_t)mt * 16 * HID;
        const unsigned short* A2 = h2prev + (size_t)mt * 16 * HID;
        v8f c = {};
        for (int k = 0; k < HID; k += 32) {
            v16bf a1 = load_a_rowmajor(A1, HID, k);
            v16bf bb = load_b_packed(Wpk_ih2, HID, k >> 1, nt * 16);
            c = wmma_bf16(a1, bb, c);
            v16bf a2 = load_a_rowmajor(A2, HID, k);
            v16bf b2f = load_b_packed(Wpk_hh2, HID, k >> 1, nt * 16);
            c = wmma_bf16(a2, b2f, c);
        }
        float bias = b2[ncol];
#pragma unroll
        for (int vi = 0; vi < 8; ++vi) {
            int row = mt * 16 + vi + 8 * lh;
            float val = tanhf(c[vi] + bias);
            h2out_bf[(size_t)row * HID + ncol] = f2bf(val);
            h2last[(size_t)row * HID + ncol] = val;
        }
    } else {
        if (!do_h1) return;
        const unsigned short* A = h1cur + (size_t)mt * 16 * HID;
        v8f c = {};
        for (int k = 0; k < HID; k += 32) {
            v16bf a = load_a_rowmajor(A, HID, k);
            v16bf bb = load_b_packed(Wpk_hh1, HID, k >> 1, nt * 16);
            c = wmma_bf16(a, bb, c);
        }
        float bias = b1[ncol];
#pragma unroll
        for (int vi = 0; vi < 8; ++vi) {
            int row = mt * 16 + vi + 8 * lh;
            float val = tanhf(c[vi] + ih1_slice[(size_t)row * HID + ncol] + bias);
            h1next_bf[(size_t)row * HID + ncol] = f2bf(val);
            h1last[(size_t)row * HID + ncol] = val;
        }
    }
}

// ---------- decoder helpers ----------
static __device__ __forceinline__ void dec_compute(
        const unsigned* lAc, const unsigned* lBc, v8f (&c)[2][4],
        int wm, int wn, int r, int lh) {
    AFQ fa[2];
    AF fb[4];
#pragma unroll
    for (int mt2 = 0; mt2 < 2; ++mt2) {
        int base = (wm * 32 + mt2 * 16 + r) * LApad + 4 * lh;
        fa[mt2].q[0] = *reinterpret_cast<const uv4*>(&lAc[base]);
        fa[mt2].q[1] = *reinterpret_cast<const uv4*>(&lAc[base + 8]);
    }
#pragma unroll
    for (int ntl = 0; ntl < 4; ++ntl)
#pragma unroll
        for (int j = 0; j < 8; ++j) {
            int kp = ((j < 4) ? j : (j + 4)) + 4 * lh;
            fb[ntl].u[j] = lBc[kp * 128 + wn * 64 + ntl * 16 + r];
        }
#pragma unroll
    for (int mt2 = 0; mt2 < 2; ++mt2)
#pragma unroll
        for (int ntl = 0; ntl < 4; ++ntl)
            c[mt2][ntl] = wmma_bf16(fa[mt2].v, fb[ntl].v, c[mt2][ntl]);
}

// ---------- kernel: decoder GEMM  logits = h2_seq @ W_dec + b_dec ----------
// 128x128 macro-tile per block; double-buffered LDS; waves in 4(M)x2(N) grid,
// each wave computes 32x64 (2 A-frags x 4 B-frags -> 8 WMMAs per chunk).
// Prepacked path uses gfx1250 async global->LDS DMA (ASYNCcnt) when available.
__global__ __launch_bounds__(256) void rnnlm_decoder(
        const unsigned short* __restrict__ h2seq,
        const unsigned* __restrict__ Bpk, const float* __restrict__ Wdec,
        int prepacked, const float* __restrict__ bdec, float* __restrict__ out) {
    __shared__ unsigned lA[2][128 * LApad];  // [buf][row][kpair(padded)]  20 KB
    __shared__ unsigned lB[2][16 * 128];     // [buf][kpair][n]            16 KB
    int bn = blockIdx.x, bm = blockIdx.y;
    int tid = threadIdx.x;
    int wv = tid >> 5, lane = tid & 31, lh = lane >> 4, r = lane & 15;
    int wm = wv >> 1, wn = wv & 1;       // wave sub-tile: rows wm*32, cols wn*64

    v8f c[2][4];
#pragma unroll
    for (int i = 0; i < 2; ++i)
#pragma unroll
        for (int j = 0; j < 4; ++j) c[i][j] = v8f{};

    const unsigned* Ag = reinterpret_cast<const unsigned*>(h2seq);  // row stride 512 u32

#if HAVE_ASYNC_LDS
    if (prepacked) {
        // Async-DMA pipeline: 512 x 16B per tile; 2 A + 2 B b128 ops per thread.
        for (int i = 0; i < 2; ++i) {
            int idx = tid + i * 256;
            int row = idx >> 2, q = idx & 3;      // A: 128 rows x 4 quads
            async_cp16(Ag + (size_t)(bm * 128 + row) * 512 + q * 4,
                       &lA[0][row * LApad + q * 4]);
            int kp = idx >> 5, nq = idx & 31;     // B: 16 kpairs x 32 quads
            async_cp16(Bpk + (size_t)kp * VOCAB + bn * 128 + nq * 4,
                       &lB[0][kp * 128 + nq * 4]);
        }
        wait_async0();
        __syncthreads();
        for (int kc = 0; kc < 32; ++kc) {
            int cur = kc & 1;
            if (kc + 1 < 32) {
                int k0 = (kc + 1) * 32;
#pragma unroll
                for (int i = 0; i < 2; ++i) {
                    int idx = tid + i * 256;
                    int row = idx >> 2, q = idx & 3;
                    async_cp16(Ag + (size_t)(bm * 128 + row) * 512 + (k0 >> 1) + q * 4,
                               &lA[cur ^ 1][row * LApad + q * 4]);
                    int kp = idx >> 5, nq = idx & 31;
                    async_cp16(Bpk + (size_t)((k0 >> 1) + kp) * VOCAB + bn * 128 + nq * 4,
                               &lB[cur ^ 1][kp * 128 + nq * 4]);
                }
            }
            dec_compute(lA[cur], lB[cur], c, wm, wn, r, lh);
            if (kc + 1 < 32) wait_async0();
            __syncthreads();
        }
    } else
#endif
    {
        unsigned aReg[8], bReg[8];
        int arow[8], akp[8], bkp[8], bnn[8];
#pragma unroll
        for (int i = 0; i < 8; ++i) {
            int idx = tid + i * 256;
            arow[i] = idx >> 4;               // A: 128 rows x 16 kpairs
            akp[i] = idx & 15;
            bkp[i] = idx >> 7;                // B: 16 kpairs x 128 cols
            bnn[i] = idx & 127;
        }
#pragma unroll
        for (int i = 0; i < 8; ++i) {
            aReg[i] = Ag[(size_t)(bm * 128 + arow[i]) * 512 + akp[i]];
            if (prepacked) {
                bReg[i] = Bpk[(size_t)bkp[i] * VOCAB + bn * 128 + bnn[i]];
            } else {
                const float* col = Wdec + (size_t)(2 * bkp[i]) * VOCAB + bn * 128 + bnn[i];
                bReg[i] = pack2(col[0], col[VOCAB]);
            }
        }
#pragma unroll
        for (int i = 0; i < 8; ++i) {
            lA[0][arow[i] * LApad + akp[i]] = aReg[i];
            lB[0][tid + i * 256] = bReg[i];
        }
        for (int kc = 0; kc < 32; ++kc) {
            __syncthreads();
            int cur = kc & 1;
            if (kc + 2 < 32) {  // prefetch chunk kc+2 (global_prefetch_b8)
                int k2 = (kc + 2) * 32;
                __builtin_prefetch(Ag + (size_t)(bm * 128 + arow[0]) * 512 + (k2 >> 1) + akp[0], 0, 0);
                if (prepacked)
                    __builtin_prefetch(Bpk + (size_t)((k2 >> 1) + bkp[0]) * VOCAB + bn * 128 + bnn[0], 0, 0);
                else
                    __builtin_prefetch(Wdec + (size_t)(k2 + 2 * bkp[0]) * VOCAB + bn * 128 + bnn[0], 0, 0);
            }
            if (kc + 1 < 32) {
                int k0 = (kc + 1) * 32;
#pragma unroll
                for (int i = 0; i < 8; ++i) {
                    aReg[i] = Ag[(size_t)(bm * 128 + arow[i]) * 512 + (k0 >> 1) + akp[i]];
                    if (prepacked) {
                        bReg[i] = Bpk[(size_t)((k0 >> 1) + bkp[i]) * VOCAB + bn * 128 + bnn[i]];
                    } else {
                        const float* col =
                            Wdec + (size_t)(k0 + 2 * bkp[i]) * VOCAB + bn * 128 + bnn[i];
                        bReg[i] = pack2(col[0], col[VOCAB]);
                    }
                }
            }
            dec_compute(lA[cur], lB[cur], c, wm, wn, r, lh);
            if (kc + 1 < 32) {
                int nxt = cur ^ 1;
#pragma unroll
                for (int i = 0; i < 8; ++i) {
                    lA[nxt][arow[i] * LApad + akp[i]] = aReg[i];
                    lB[nxt][tid + i * 256] = bReg[i];
                }
            }
        }
    }

#pragma unroll
    for (int mt2 = 0; mt2 < 2; ++mt2)
#pragma unroll
        for (int ntl = 0; ntl < 4; ++ntl) {
            int gcol = bn * 128 + wn * 64 + ntl * 16 + r;
            float bias = bdec[gcol];
#pragma unroll
            for (int vi = 0; vi < 8; ++vi) {
                int grow = bm * 128 + wm * 32 + mt2 * 16 + vi + 8 * lh;
                out[(size_t)grow * VOCAB + gcol] = c[mt2][ntl][vi] + bias;
            }
        }
}

// ---------- kernel: write hidden_out tail ----------
__global__ void rnnlm_tail(const float* __restrict__ h1last,
                           const float* __restrict__ h2last,
                           float* __restrict__ out) {
    int i = blockIdx.x * blockDim.x + threadIdx.x;  // 65536 total
    if (i < BATCH * HID) out[i] = h1last[i];
    else out[i] = h2last[i - BATCH * HID];
}

// ---------- host ----------
extern "C" void kernel_launch(void* const* d_in, const int* in_sizes, int n_in,
                              void* d_out, int out_size, void* d_ws, size_t ws_size,
                              hipStream_t stream) {
    (void)in_sizes; (void)n_in; (void)out_size;
    const int*   x     = (const int*)  d_in[0];
    const float* hid   = (const float*)d_in[1];   // (2,32,1024)
    const float* emb   = (const float*)d_in[2];
    const float* W_ih1 = (const float*)d_in[3];
    const float* W_hh1 = (const float*)d_in[4];
    const float* b1    = (const float*)d_in[5];
    const float* W_ih2 = (const float*)d_in[6];
    const float* W_hh2 = (const float*)d_in[7];
    const float* b2    = (const float*)d_in[8];
    const float* W_dec = (const float*)d_in[9];
    const float* b_dec = (const float*)d_in[10];
    float* outp = (float*)d_out;

    char* ws = (char*)d_ws;
    const size_t MB = 1u << 20;
    unsigned* pk_ih1 = (unsigned*)(ws + 0 * MB);          // 2 MB each
    unsigned* pk_hh1 = (unsigned*)(ws + 2 * MB);
    unsigned* pk_ih2 = (unsigned*)(ws + 4 * MB);
    unsigned* pk_hh2 = (unsigned*)(ws + 6 * MB);
    float*    ih1all = (float*)   (ws + 8 * MB);          // 16 MB
    unsigned short* h2seq  = (unsigned short*)(ws + 24 * MB); // 8 MB
    unsigned short* hid_bf = (unsigned short*)(ws + 32 * MB); // 128 KB
    unsigned short* h1b0   = hid_bf + 2 * BATCH * HID;        // 64 KB each
    unsigned short* h1b1   = h1b0 + BATCH * HID;
    float* h1last = (float*)((char*)(h1b1 + BATCH * HID));
    float* h2last = h1last + BATCH * HID;
    unsigned* pk_dec = (unsigned*)(ws + 36 * MB);         // 62.5 MB (optional)
    size_t need_dec = 36 * MB + (size_t)(HID / 2) * VOCAB * 4;
    int prepacked = (ws_size >= need_dec) ? 1 : 0;

    // 1) pack weights to pair-interleaved bf16
    rnnlm_pack_w<<<2048, 256, 0, stream>>>(W_ih1, pk_ih1, HID, HID);
    rnnlm_pack_w<<<2048, 256, 0, stream>>>(W_hh1, pk_hh1, HID, HID);
    rnnlm_pack_w<<<2048, 256, 0, stream>>>(W_ih2, pk_ih2, HID, HID);
    rnnlm_pack_w<<<2048, 256, 0, stream>>>(W_hh2, pk_hh2, HID, HID);
    if (prepacked)
        rnnlm_pack_w<<<8192, 256, 0, stream>>>(W_dec, pk_dec, HID, VOCAB);
    // 2) initial hidden -> bf16
    rnnlm_cvt_bf16<<<256, 256, 0, stream>>>(hid, hid_bf, 2 * BATCH * HID);
    // 3) ih1_all = gather(emb,x) @ W_ih1
    rnnlm_gemm_ih1<<<2048, 256, 0, stream>>>(x, emb, pk_ih1, ih1all);

    // 4) recurrence
    unsigned short* h1buf[2] = {h1b0, h1b1};
    rnnlm_step<<<32, 256, 0, stream>>>(
        0, 1, hid_bf, (const unsigned short*)nullptr,
        pk_hh1, pk_ih2, pk_hh2, ih1all, b1, b2,
        h1buf[0], h1last, (unsigned short*)nullptr, (float*)nullptr);
    for (int t = 0; t < SEQ; ++t) {
        const unsigned short* h1c = h1buf[t & 1];
        const unsigned short* h2p =
            (t == 0) ? (hid_bf + BATCH * HID) : (h2seq + (size_t)(t - 1) * BATCH * HID);
        int do_h1 = (t < SEQ - 1);
        rnnlm_step<<<32, 256, 0, stream>>>(
            1, do_h1, h1c, h2p, pk_hh1, pk_ih2, pk_hh2,
            ih1all + (size_t)(t + 1) * BATCH * HID, b1, b2,
            h1buf[(t + 1) & 1], h1last,
            h2seq + (size_t)t * BATCH * HID, h2last);
    }

    // 5) decoder GEMM -> logits
    rnnlm_decoder<<<dim3(VOCAB / 128, MTOT / 128), 256, 0, stream>>>(
        h2seq, pk_dec, W_dec, prepacked, b_dec, outp);
    // 6) hidden_out tail
    rnnlm_tail<<<(2 * BATCH * HID) / 256, 256, 0, stream>>>(
        h1last, h2last, outp + (size_t)MTOT * VOCAB);
}
